// BERTLanguageModelWithSFTMask_3633542333025
// MI455X (gfx1250) — matrix operand order repaired and checked by reading, not compiled, not run on hardware
//
#include <hip/hip_runtime.h>
#include <hip/hip_bf16.h>
#include <math.h>

// Problem constants (match reference)
#define B_  2
#define S_  1024
#define H_  1024
#define L_  4
#define NH_ 16
#define DH_ 64
#define FF_ 4096
#define V_  32000
#define T_  (B_ * S_)   // 2048 tokens

typedef unsigned short u16;
typedef __attribute__((ext_vector_type(16))) __bf16          v16bf;
typedef __attribute__((ext_vector_type(8)))  float           v8f;
typedef __attribute__((ext_vector_type(8)))  unsigned short  u16x8;

static __device__ __forceinline__ u16 f2bf(float f) {
  unsigned u = __float_as_uint(f);
  u = u + 0x7FFFu + ((u >> 16) & 1u);   // round-to-nearest-even
  return (u16)(u >> 16);
}

union Frag16 { v16bf v; u16x8 h[2]; };

// Async global->LDS copy of 32 bytes per lane (two b128 transfers), GV mode.
// The immediate offset is added to both the LDS and the global address.
static __device__ __forceinline__ void async_copy_b32bytes(unsigned ldsOff,
                                                           const void* gaddr) {
  asm volatile("global_load_async_to_lds_b128 %0, %1, off"
               :: "v"(ldsOff), "v"(gaddr) : "memory");
  asm volatile("global_load_async_to_lds_b128 %0, %1, off offset:16"
               :: "v"(ldsOff), "v"(gaddr) : "memory");
}
static __device__ __forceinline__ void async_wait0() {
  asm volatile("s_wait_asynccnt 0x0" ::: "memory");
}

// ---------------------------------------------------------------------------
// fp32 -> bf16 conversion WITH transpose: src [R][C] f32 -> dst [C][R] bf16.
// blockIdx.z selects a matrix in a contiguous stack (layer index).
// 32x32 LDS tile, 256 threads (32x8): coalesced read and coalesced write.
// ---------------------------------------------------------------------------
__global__ __launch_bounds__(256) void k_cvt_bf16_t(
    const float* __restrict__ src, u16* __restrict__ dst, int R, int C) {
  __shared__ float tile[32][33];
  const long mofs = (long)blockIdx.z * R * C;
  const int r0 = blockIdx.y * 32, c0 = blockIdx.x * 32;
  const int tx = threadIdx.x & 31, ty = threadIdx.x >> 5;   // 32 x 8
#pragma unroll
  for (int i = 0; i < 32; i += 8) {
    const int r = r0 + ty + i, c = c0 + tx;
    tile[ty + i][tx] = (r < R && c < C) ? src[mofs + (long)r * C + c] : 0.f;
  }
  __syncthreads();
#pragma unroll
  for (int i = 0; i < 32; i += 8) {
    const int c = c0 + ty + i, r = r0 + tx;
    if (c < C && r < R) dst[mofs + (long)c * R + r] = f2bf(tile[tx][ty + i]);
  }
}

// ---------------------------------------------------------------------------
// Embedding gather + LayerNorm  (one block per token, H=1024, 4 elems/thread)
// ---------------------------------------------------------------------------
__global__ __launch_bounds__(256) void k_embed_ln(
    const int* __restrict__ x, const float* __restrict__ wemb,
    const float* __restrict__ pemb, const float* __restrict__ temb,
    const float* __restrict__ gam, const float* __restrict__ bet,
    float* __restrict__ hf, u16* __restrict__ hb) {
  __shared__ float red[256];
  const int t   = blockIdx.x;
  const int s   = t & (S_ - 1);
  const int tok = x[t];
  const int tid = threadIdx.x;
  float v[4];
  float sum = 0.f;
#pragma unroll
  for (int j = 0; j < 4; j++) {
    const int c = tid + j * 256;
    v[j] = wemb[(long)tok * H_ + c] + pemb[(long)s * H_ + c] + temb[c];
    sum += v[j];
  }
  red[tid] = sum; __syncthreads();
  for (int st = 128; st > 0; st >>= 1) { if (tid < st) red[tid] += red[tid + st]; __syncthreads(); }
  const float mean = red[0] * (1.f / H_); __syncthreads();
  float sq = 0.f;
#pragma unroll
  for (int j = 0; j < 4; j++) { const float d = v[j] - mean; sq += d * d; }
  red[tid] = sq; __syncthreads();
  for (int st = 128; st > 0; st >>= 1) { if (tid < st) red[tid] += red[tid + st]; __syncthreads(); }
  const float rstd = rsqrtf(red[0] * (1.f / H_) + 1e-12f);
#pragma unroll
  for (int j = 0; j < 4; j++) {
    const int c = tid + j * 256;
    const float y = (v[j] - mean) * rstd * gam[c] + bet[c];
    hf[(long)t * H_ + c] = y;
    hb[(long)t * H_ + c] = f2bf(y);
  }
}

// ---------------------------------------------------------------------------
// LayerNorm over fp32 rows -> fp32 + bf16
// ---------------------------------------------------------------------------
__global__ __launch_bounds__(256) void k_layernorm(
    const float* __restrict__ xin, const float* __restrict__ gam,
    const float* __restrict__ bet, float* __restrict__ hf, u16* __restrict__ hb) {
  __shared__ float red[256];
  const int t = blockIdx.x;
  const int tid = threadIdx.x;
  const float* p = xin + (long)t * H_;
  float v[4];
  float sum = 0.f;
#pragma unroll
  for (int j = 0; j < 4; j++) { v[j] = p[tid + j * 256]; sum += v[j]; }
  red[tid] = sum; __syncthreads();
  for (int st = 128; st > 0; st >>= 1) { if (tid < st) red[tid] += red[tid + st]; __syncthreads(); }
  const float mean = red[0] * (1.f / H_); __syncthreads();
  float sq = 0.f;
#pragma unroll
  for (int j = 0; j < 4; j++) { const float d = v[j] - mean; sq += d * d; }
  red[tid] = sq; __syncthreads();
  for (int st = 128; st > 0; st >>= 1) { if (tid < st) red[tid] += red[tid + st]; __syncthreads(); }
  const float rstd = rsqrtf(red[0] * (1.f / H_) + 1e-12f);
#pragma unroll
  for (int j = 0; j < 4; j++) {
    const int c = tid + j * 256;
    const float y = (v[j] - mean) * rstd * gam[c] + bet[c];
    hf[(long)t * H_ + c] = y;
    hb[(long)t * H_ + c] = f2bf(y);
  }
}

// ---------------------------------------------------------------------------
// Masked attention softmax: scores fp32 [B,NH,S,S] -> probs bf16
// One block per (b, head, q) row.  scale = 1/sqrt(DH), causal + padding mask.
// ---------------------------------------------------------------------------
__global__ __launch_bounds__(256) void k_attn_softmax(
    const float* __restrict__ scores, u16* __restrict__ probs,
    const float* __restrict__ amask) {
  __shared__ float red[256];
  const int row = blockIdx.x;           // b*NH*S + h*S + q
  const int q   = row & (S_ - 1);
  const int b   = row >> 14;            // / (NH_*S_) == / 16384
  const int tid = threadIdx.x;
  const float scale = 0.125f;           // 1/sqrt(64)
  const float* src = scores + (long)row * S_;
  float vals[4];
  float mx = -3.4e38f;
#pragma unroll
  for (int j = 0; j < 4; j++) {
    const int k = tid + j * 256;
    const float keep = (k <= q) ? amask[b * S_ + k] : 0.f;
    const float v = src[k] * scale + (1.f - keep) * (-1e9f);
    vals[j] = v; mx = fmaxf(mx, v);
  }
  red[tid] = mx; __syncthreads();
  for (int st = 128; st > 0; st >>= 1) { if (tid < st) red[tid] = fmaxf(red[tid], red[tid + st]); __syncthreads(); }
  const float rmax = red[0]; __syncthreads();
  float sum = 0.f;
#pragma unroll
  for (int j = 0; j < 4; j++) { vals[j] = __expf(vals[j] - rmax); sum += vals[j]; }
  red[tid] = sum; __syncthreads();
  for (int st = 128; st > 0; st >>= 1) { if (tid < st) red[tid] += red[tid + st]; __syncthreads(); }
  const float rs = 1.f / red[0];
  u16* dst = probs + (long)row * S_;
#pragma unroll
  for (int j = 0; j < 4; j++) dst[tid + j * 256] = f2bf(vals[j] * rs);
}

// ---------------------------------------------------------------------------
// In-place vocab softmax over V=32000 per token row
// ---------------------------------------------------------------------------
__global__ __launch_bounds__(256) void k_vocab_softmax(float* __restrict__ logits) {
  __shared__ float red[256];
  const int t = blockIdx.x;
  const int tid = threadIdx.x;
  float* p = logits + (long)t * V_;
  float mx = -3.4e38f;
  for (int j = tid; j < V_; j += 256) mx = fmaxf(mx, p[j]);
  red[tid] = mx; __syncthreads();
  for (int st = 128; st > 0; st >>= 1) { if (tid < st) red[tid] = fmaxf(red[tid], red[tid + st]); __syncthreads(); }
  mx = red[0]; __syncthreads();
  float sum = 0.f;
  for (int j = tid; j < V_; j += 256) sum += __expf(p[j] - mx);
  red[tid] = sum; __syncthreads();
  for (int st = 128; st > 0; st >>= 1) { if (tid < st) red[tid] += red[tid + st]; __syncthreads(); }
  const float rs = 1.f / red[0]; __syncthreads();
  for (int j = tid; j < V_; j += 256) p[j] = __expf(p[j] - mx) * rs;
}

// ---------------------------------------------------------------------------
// Generic batched bf16 WMMA GEMM:  C[M,N] = A[M,K] * op(B)[K,N] (+bias)(+res)(gelu)
// Block tile 128x128, 8 waves (2x4), wave tile 64x32 = 4x2 WMMA 16x16x32 tiles.
// Double-buffered LDS (2 x 20KB): stage tile k+1 while computing tile k ->
// one barrier per K-step.  Fast-path staging uses GLOBAL_LOAD_ASYNC_TO_LDS_B128
// (ASYNCcnt-tracked LDS DMA, no VGPR round trip); s_wait_asynccnt 0 precedes
// the barrier.  sA row-major [m][k], sB transposed [n][k]; pitch 40 (80B rows:
// 16B-aligned, conflict-free for 16-lane row groups).  Fragments are two
// ds_load_b128 each, matching ISA 7.12.2 bf16 VGPR layouts.
// ---------------------------------------------------------------------------
struct GemmArgs {
  const u16* A; const u16* Bm; const float* bias; const float* residual;
  float* Cf; u16* Cbf;
  int M, N, K, lda, ldb, ldc;
  long aOuter, aInner, bOuter, bInner, cOuter, cInner;
  int innerCount; int transB; int gelu;
};

__global__ __launch_bounds__(256) void k_gemm_bf16(GemmArgs g) {
  __shared__ __align__(16) u16 sA[2][128][40];
  __shared__ __align__(16) u16 sB[2][128][40];
  const int tid  = threadIdx.x;
  const int lane = tid & 31;
  const int wid  = tid >> 5;
  const int m0 = (wid >> 2) * 64;   // wave row origin within block
  const int n0 = (wid & 3) * 32;    // wave col origin within block
  const int bm = blockIdx.y * 128;
  const int bn = blockIdx.x * 128;
  const int z  = blockIdx.z;
  const long za = (long)(z / g.innerCount) * g.aOuter + (long)(z % g.innerCount) * g.aInner;
  const long zb = (long)(z / g.innerCount) * g.bOuter + (long)(z % g.innerCount) * g.bInner;
  const long zc = (long)(z / g.innerCount) * g.cOuter + (long)(z % g.innerCount) * g.cInner;
  const u16* __restrict__ Ag = g.A  + za;
  const u16* __restrict__ Bg = g.Bm + zb;

  // Stage one 128x32 K-tile of A and B into LDS buffer `buf`.
  auto stage = [&](int kt, int buf) {
    // ---- A tile: each thread owns one (m, 16-wide k half)
    {
      const int m  = tid >> 1;
      const int kh = (tid & 1) * 16;
      const int grow = bm + m, gk = kt + kh;
      if (grow < g.M && (gk + 15) < g.K) {
        const u16* src = Ag + (long)grow * g.lda + gk;
        async_copy_b32bytes((unsigned)(size_t)&sA[buf][m][kh], src);
        if (gk + 47 < g.K) __builtin_prefetch(src + 32, 0, 1);  // next K-tile
      } else {
#pragma unroll
        for (int i = 0; i < 16; i++)
          sA[buf][m][kh + i] = (grow < g.M && (gk + i) < g.K)
                                   ? Ag[(long)grow * g.lda + gk + i] : (u16)0;
      }
    }
    // ---- B tile into sB[n][k]
    if (g.transB) {  // B stored [N,K] row-major: contiguous in k (all weights)
      const int n  = tid >> 1;
      const int kh = (tid & 1) * 16;
      const int gcol = bn + n, gk = kt + kh;
      if (gcol < g.N && (gk + 15) < g.K) {
        const u16* src = Bg + (long)gcol * g.ldb + gk;
        async_copy_b32bytes((unsigned)(size_t)&sB[buf][n][kh], src);
        if (gk + 47 < g.K) __builtin_prefetch(src + 32, 0, 1);
      } else {
#pragma unroll
        for (int i = 0; i < 16; i++)
          sB[buf][n][kh + i] = (gcol < g.N && (gk + i) < g.K)
                                   ? Bg[(long)gcol * g.ldb + gk + i] : (u16)0;
      }
    } else {         // B stored [K,N] row-major: contiguous in n, LDS transpose
      const int k  = tid >> 3;
      const int nq = (tid & 7) * 16;
      const int gk = kt + k;
      if (gk < g.K && (bn + nq + 15) < g.N) {
        const u16x8* src = (const u16x8*)(Bg + (long)gk * g.ldb + bn + nq);
        const u16x8 d0 = src[0], d1 = src[1];
#pragma unroll
        for (int i = 0; i < 8; i++) sB[buf][nq + i][k] = d0[i];
#pragma unroll
        for (int i = 0; i < 8; i++) sB[buf][nq + 8 + i][k] = d1[i];
        if (gk + 32 < g.K)
          __builtin_prefetch(Bg + (long)(gk + 32) * g.ldb + bn + nq, 0, 1);
      } else {
#pragma unroll
        for (int i = 0; i < 16; i++) {
          const int gcol = bn + nq + i;
          sB[buf][nq + i][k] = (gcol < g.N && gk < g.K)
                                   ? Bg[(long)gk * g.ldb + gcol] : (u16)0;
        }
      }
    }
  };

  v8f acc[4][2];
#pragma unroll
  for (int i = 0; i < 4; i++)
#pragma unroll
    for (int j = 0; j < 2; j++)
#pragma unroll
      for (int e = 0; e < 8; e++) acc[i][j][e] = 0.f;

  // Fragment geometry (ISA 7.12.2 bf16 layouts)
  const int rA  = lane & 15;
  const int kbA = (lane < 16) ? 0 : 8;   // lanes 0-15: K{0..7,16..23}; 16-31: K{8..15,24..31}
  const int nB  = lane & 15;
  const int kbB = (lane >> 4) * 8;       // lanes 0-15: K{0..7,16..23}; 16-31: K{8..15,24..31}

  stage(0, 0);
  async_wait0();
  __syncthreads();
  int ib = 0;
  for (int kt = 0; kt < g.K; kt += 32) {
    if (kt + 32 < g.K) stage(kt + 32, ib ^ 1);   // fill other buffer (async)

    Frag16 fa[4], fb[2];
#pragma unroll
    for (int mi = 0; mi < 4; mi++) {
      const int row = m0 + mi * 16 + rA;
      fa[mi].h[0] = *(const u16x8*)&sA[ib][row][kbA];
      fa[mi].h[1] = *(const u16x8*)&sA[ib][row][kbA + 16];
    }
#pragma unroll
    for (int ni = 0; ni < 2; ni++) {
      const int col = n0 + ni * 16 + nB;
      fb[ni].h[0] = *(const u16x8*)&sB[ib][col][kbB];
      fb[ni].h[1] = *(const u16x8*)&sB[ib][col][kbB + 16];
    }
#pragma unroll
    for (int mi = 0; mi < 4; mi++)
#pragma unroll
      for (int ni = 0; ni < 2; ni++)
        acc[mi][ni] = __builtin_amdgcn_wmma_f32_16x16x32_bf16(
            false, fa[mi].v, false, fb[ni].v, (short)0, acc[mi][ni], false, false);
    async_wait0();        // inline-asm async loads are untracked by compiler
    __syncthreads();
    ib ^= 1;
  }

  // ---- epilogue: bias / residual / GELU / fp32 and-or bf16 stores
  const int g16 = lane >> 4;
#pragma unroll
  for (int mi = 0; mi < 4; mi++) {
#pragma unroll
    for (int ni = 0; ni < 2; ni++) {
      const int col = bn + n0 + ni * 16 + (lane & 15);
      if (col >= g.N) continue;
      const float badd = g.bias ? g.bias[col] : 0.f;
#pragma unroll
      for (int i = 0; i < 8; i++) {
        const int row = bm + m0 + mi * 16 + g16 * 8 + i;  // C/D: M = i + 8*(lane>=16)
        if (row >= g.M) continue;
        const long off = zc + (long)row * g.ldc + col;
        float v = acc[mi][ni][i] + badd;
        if (g.residual) v += g.residual[off];
        if (g.gelu)     v = 0.5f * v * (1.f + erff(v * 0.70710678118654752f));
        if (g.Cf)  g.Cf[off]  = v;
        if (g.Cbf) g.Cbf[off] = f2bf(v);
      }
    }
  }
}

// ---------------------------------------------------------------------------
// Host orchestration
// ---------------------------------------------------------------------------
static inline size_t alignup256(size_t v) { return (v + 255) & ~(size_t)255; }

extern "C" void kernel_launch(void* const* d_in, const int* in_sizes, int n_in,
                              void* d_out, int out_size, void* d_ws, size_t ws_size,
                              hipStream_t stream) {
  (void)in_sizes; (void)n_in; (void)out_size; (void)ws_size;
  // Input order: x, attention_mask, then params in dict order
  const int*   x        = (const int*)  d_in[0];
  const float* amask    = (const float*)d_in[1];
  const float* word_emb = (const float*)d_in[2];
  const float* pos_emb  = (const float*)d_in[3];
  const float* tok_emb  = (const float*)d_in[4];
  const float* emb_ln_g = (const float*)d_in[5];
  const float* emb_ln_b = (const float*)d_in[6];
  const float* Wq = (const float*)d_in[7];   const float* bq = (const float*)d_in[8];
  const float* Wk = (const float*)d_in[9];   const float* bk = (const float*)d_in[10];
  const float* Wv = (const float*)d_in[11];  const float* bv = (const float*)d_in[12];
  const float* Wo = (const float*)d_in[13];  const float* bo = (const float*)d_in[14];
  const float* ln1_g = (const float*)d_in[15]; const float* ln1_b = (const float*)d_in[16];
  const float* Wi = (const float*)d_in[17];  const float* bi = (const float*)d_in[18];
  const float* Wf = (const float*)d_in[19];  const float* bfp = (const float*)d_in[20];
  const float* ln2_g = (const float*)d_in[21]; const float* ln2_b = (const float*)d_in[22];
  const float* lm_w = (const float*)d_in[23];  const float* lm_b = (const float*)d_in[24];

  // Workspace carve-up
  char* w = (char*)d_ws;
  size_t off = 0;
  auto take = [&](size_t bytes) { void* p = w + off; off = alignup256(off + bytes); return p; };
  float* h_f32   = (float*)take((size_t)T_ * H_ * 4);
  u16*   h_bf    = (u16*)  take((size_t)T_ * H_ * 2);
  u16*   qb      = (u16*)  take((size_t)T_ * H_ * 2);
  u16*   kb      = (u16*)  take((size_t)T_ * H_ * 2);
  u16*   vb      = (u16*)  take((size_t)T_ * H_ * 2);
  u16*   ctx_bf  = (u16*)  take((size_t)T_ * H_ * 2);
  float* tmp_f32 = (float*)take((size_t)T_ * H_ * 4);
  u16*   ff_bf   = (u16*)  take((size_t)T_ * FF_ * 2);
  float* scores  = (float*)take((size_t)B_ * NH_ * S_ * S_ * 4);
  u16*   probs   = (u16*)  take((size_t)B_ * NH_ * S_ * S_ * 2);
  u16*   wq_bf   = (u16*)  take((size_t)L_ * H_ * H_ * 2);   // transposed [H][H] per layer
  u16*   wk_bf   = (u16*)  take((size_t)L_ * H_ * H_ * 2);
  u16*   wv_bf   = (u16*)  take((size_t)L_ * H_ * H_ * 2);
  u16*   wo_bf   = (u16*)  take((size_t)L_ * H_ * H_ * 2);
  u16*   wi_bf   = (u16*)  take((size_t)L_ * H_ * FF_ * 2);  // transposed [FF][H] per layer
  u16*   wf_bf   = (u16*)  take((size_t)L_ * FF_ * H_ * 2);  // transposed [H][FF] per layer
  u16*   lmw_bf  = (u16*)  take((size_t)H_ * V_ * 2);        // transposed [V][H]

  // Weight conversion fp32 -> bf16 with transpose (per call; deterministic).
  {
    dim3 gHH(H_ / 32, H_ / 32, L_);
    k_cvt_bf16_t<<<gHH, 256, 0, stream>>>(Wq, wq_bf, H_, H_);
    k_cvt_bf16_t<<<gHH, 256, 0, stream>>>(Wk, wk_bf, H_, H_);
    k_cvt_bf16_t<<<gHH, 256, 0, stream>>>(Wv, wv_bf, H_, H_);
    k_cvt_bf16_t<<<gHH, 256, 0, stream>>>(Wo, wo_bf, H_, H_);
    dim3 gHF(FF_ / 32, H_ / 32, L_);
    k_cvt_bf16_t<<<gHF, 256, 0, stream>>>(Wi, wi_bf, H_, FF_);
    dim3 gFH(H_ / 32, FF_ / 32, L_);
    k_cvt_bf16_t<<<gFH, 256, 0, stream>>>(Wf, wf_bf, FF_, H_);
    dim3 gHV(V_ / 32, H_ / 32, 1);
    k_cvt_bf16_t<<<gHV, 256, 0, stream>>>(lm_w, lmw_bf, H_, V_);
  }

  // Embeddings + LN
  k_embed_ln<<<T_, 256, 0, stream>>>(x, word_emb, pos_emb, tok_emb,
                                     emb_ln_g, emb_ln_b, h_f32, h_bf);

  auto gemm = [&](const u16* A, int lda, long aO, long aI,
                  const u16* Bm, int ldb, long bO, long bI, int transB,
                  const float* bias, const float* resid,
                  float* Cf, u16* Cbf, int ldc, long cO, long cI,
                  int M, int N, int K, int batch, int innerCount, int gelu) {
    GemmArgs ga;
    ga.A = A; ga.Bm = Bm; ga.bias = bias; ga.residual = resid;
    ga.Cf = Cf; ga.Cbf = Cbf;
    ga.M = M; ga.N = N; ga.K = K; ga.lda = lda; ga.ldb = ldb; ga.ldc = ldc;
    ga.aOuter = aO; ga.aInner = aI; ga.bOuter = bO; ga.bInner = bI;
    ga.cOuter = cO; ga.cInner = cI;
    ga.innerCount = innerCount; ga.transB = transB; ga.gelu = gelu;
    dim3 grid((N + 127) / 128, (M + 127) / 128, batch);
    k_gemm_bf16<<<grid, 256, 0, stream>>>(ga);
  };

  for (int l = 0; l < L_; l++) {
    const long wHH = (long)l * H_ * H_;
    // Q, K, V projections (transposed weights -> transB=1 async staging)
    gemm(h_bf, H_, 0, 0, wq_bf + wHH, H_, 0, 0, 1, bq + l * H_, nullptr,
         nullptr, qb, H_, 0, 0, T_, H_, H_, 1, 1, 0);
    gemm(h_bf, H_, 0, 0, wk_bf + wHH, H_, 0, 0, 1, bk + l * H_, nullptr,
         nullptr, kb, H_, 0, 0, T_, H_, H_, 1, 1, 0);
    gemm(h_bf, H_, 0, 0, wv_bf + wHH, H_, 0, 0, 1, bv + l * H_, nullptr,
         nullptr, vb, H_, 0, 0, T_, H_, H_, 1, 1, 0);
    // scores[b,h] = Q_slice @ K_slice^T   (batch = B*NH, per-head strides)
    gemm(qb, H_, (long)S_ * H_, DH_,
         kb, H_, (long)S_ * H_, DH_, /*transB=*/1,
         nullptr, nullptr, scores, nullptr, S_,
         (long)NH_ * S_ * S_, (long)S_ * S_,
         S_, S_, DH_, B_ * NH_, NH_, 0);
    // masked softmax -> bf16 probs
    k_attn_softmax<<<B_ * NH_ * S_, 256, 0, stream>>>(scores, probs, amask);
    // ctx[b,h] = probs @ V_slice
    gemm(probs, S_, (long)NH_ * S_ * S_, (long)S_ * S_,
         vb, H_, (long)S_ * H_, DH_, /*transB=*/0,
         nullptr, nullptr, nullptr, ctx_bf, H_,
         (long)S_ * H_, DH_,
         S_, DH_, S_, B_ * NH_, NH_, 0);
    // attn out: ctx @ Wo + bo + h  -> tmp; LN1 -> h
    gemm(ctx_bf, H_, 0, 0, wo_bf + wHH, H_, 0, 0, 1, bo + l * H_, h_f32,
         tmp_f32, nullptr, H_, 0, 0, T_, H_, H_, 1, 1, 0);
    k_layernorm<<<T_, 256, 0, stream>>>(tmp_f32, ln1_g + l * H_, ln1_b + l * H_,
                                        h_f32, h_bf);
    // FFN up + exact GELU -> bf16   (wi_bf is [FF][H] transposed, ldb=H)
    gemm(h_bf, H_, 0, 0, wi_bf + (long)l * H_ * FF_, H_, 0, 0, 1, bi + l * FF_,
         nullptr, nullptr, ff_bf, FF_, 0, 0, T_, FF_, H_, 1, 1, /*gelu=*/1);
    // FFN down + residual -> tmp; LN2 -> h   (wf_bf is [H][FF] transposed, ldb=FF)
    gemm(ff_bf, FF_, 0, 0, wf_bf + (long)l * FF_ * H_, FF_, 0, 0, 1, bfp + l * H_,
         h_f32, tmp_f32, nullptr, H_, 0, 0, T_, H_, FF_, 1, 1, 0);
    k_layernorm<<<T_, 256, 0, stream>>>(tmp_f32, ln2_g + l * H_, ln2_b + l * H_,
                                        h_f32, h_bf);
  }

  // LM head -> logits (fp32, directly into d_out), then in-place softmax
  // (lmw_bf is [V][H] transposed, ldb=H)
  gemm(h_bf, H_, 0, 0, lmw_bf, H_, 0, 0, 1, lm_b, nullptr,
       (float*)d_out, nullptr, V_, 0, 0, T_, V_, H_, 1, 1, 0);
  k_vocab_softmax<<<T_, 256, 0, stream>>>((float*)d_out);
}
